// PointEstimator_6493990552316
// MI455X (gfx1250) — matrix operand order repaired and checked by reading, not compiled, not run on hardware
//
#include <hip/hip_runtime.h>

// ---------------------------------------------------------------------------
// PointEstimator for MI455X (gfx1250, wave32, WMMA).
//
// Stage 1 (dominant): logits^T GEMM via V_WMMA_F32_16X16X32_BF16 with bf16x3
//   fp32 emulation, fused per-lane argmax over the M (output) dimension.
//   Blocking: block = one point-head p x 64 batch rows (4 waves x 16), so each
//   block touches only its 688 KB W_p slice -> ~700 MB total L2 traffic
//   instead of 2.8 GB with batch-major blocking. Two M-tiles per K-chunk give
//   two independent accumulator chains and halve LDS B reads.
// Stage 2: tiny params GEMM in plain fp32.
// Stage 3: slot(t) = #{i: pts[i] <= t}  (sort-free), linear compose.
// ---------------------------------------------------------------------------

typedef __attribute__((ext_vector_type(16))) __bf16 v16bf;
typedef __attribute__((ext_vector_type(8)))  float  v8f;

#define SEQ_LEN   512
#define PRED_LEN  336
#define N_PTS     8
#define N_SLOTS   9
#define BATCH     8192
#define KCHUNKS   16            // 512 / 32
#define MT_PER_P  21            // 336 / 16
#define NT_TOTAL  168           // 8 * 21
#define W_SW_ELEMS (NT_TOTAL * KCHUNKS * 32 * 16)   // 1,376,256 bf16 per array
#define BR        64            // batch rows per block
#define GEMM_THREADS 128        // 4 waves = 4 batch subtiles

// ---------------------------------------------------------------------------
// Pre-swizzle W_pts into the 16-bit A-matrix VGPR layout (ISA 7.12.2):
//  lane l (m = l&15, half = l>>4), v16bf element e:
//    e<8 : k = kc*32 + half*8 + e          (VGPR0-3)
//    e>=8: k = kc*32 + 16 + half*8 + (e-8) (VGPR4-7)
// Flat index: (((gt*16 + kc)*32) + l)*16 + e, gt = global 16-row tile (0..167)
// ---------------------------------------------------------------------------
__global__ __launch_bounds__(256) void prep_w_kernel(
    const float* __restrict__ Wp,
    __bf16* __restrict__ Whi, __bf16* __restrict__ Wlo)
{
    int i = blockIdx.x * 256 + threadIdx.x;
    if (i >= W_SW_ELEMS) return;
    int e    = i & 15;
    int l    = (i >> 4) & 31;
    int kc   = (i >> 9) & 15;
    int gt   = i >> 13;
    int m    = l & 15;
    int half = l >> 4;
    int k    = kc * 32 + ((e >= 8) ? 16 : 0) + half * 8 + (e & 7);
    int row  = gt * 16 + m;
    float w  = Wp[row * SEQ_LEN + k];
    __bf16 hi = (__bf16)w;
    __bf16 lo = (__bf16)(w - (float)hi);
    Whi[i] = hi;
    Wlo[i] = lo;
}

// ---------------------------------------------------------------------------
// GEMM + fused argmax. Grid (BATCH/BR, N_PTS); block = 4 waves, wave w owns a
// 16-row batch subtile and all 21 M-tiles of head p = blockIdx.y.
// xc bf16 hi/lo in LDS, 16-bit B-matrix layout: lane l (n=l&15, half=l>>4),
// element e -> k = kc*32 + (e>=8 ? 16 : 0) + half*8 + (e&7).
// C/D layout: lane (n=lane&15, half), VGPR v -> M row = half*8 + v, so the
// argmax over outputs is lane-local + one __shfl_xor(16) cross-half combine.
// ---------------------------------------------------------------------------
__global__ __launch_bounds__(GEMM_THREADS) void gemm_argmax_kernel(
    const float*  __restrict__ x,
    const __bf16* __restrict__ Whi,
    const __bf16* __restrict__ Wlo,
    const float*  __restrict__ bpts,
    int* __restrict__ pts)
{
    __shared__ __bf16 sXhi[BR * SEQ_LEN];   // 64 KB
    __shared__ __bf16 sXlo[BR * SEQ_LEN];   // 64 KB
    __shared__ float  sLast[BR];

    const int bg  = blockIdx.x;        // batch group (BR rows)
    const int p   = blockIdx.y;        // point head
    const int tid = threadIdx.x;

    if (tid < BR) sLast[tid] = x[(bg * BR + tid) * SEQ_LEN + (SEQ_LEN - 1)];
    __syncthreads();

    // Fill LDS with bf16 hi/lo of xc, layout [sub(4)][kc(16)][lane(32)][e(16)].
    for (int i = tid; i < BR * SEQ_LEN; i += GEMM_THREADS) {
        int e    = i & 15;
        int l    = (i >> 4) & 31;
        int kc   = (i >> 9) & 15;
        int sub  = i >> 13;
        int bl   = sub * 16 + (l & 15);
        int half = l >> 4;
        int k    = kc * 32 + ((e >= 8) ? 16 : 0) + half * 8 + (e & 7);
        float xv = x[(bg * BR + bl) * SEQ_LEN + k] - sLast[bl];
        __bf16 hi = (__bf16)xv;
        sXhi[i] = hi;
        sXlo[i] = (__bf16)(xv - (float)hi);
    }
    __syncthreads();

    const int wv   = tid >> 5;         // batch subtile 0..3
    const int lane = tid & 31;
    const int half = lane >> 4;

    const v16bf* sBhi = (const v16bf*)sXhi;   // [sub*512 + kc*32 + lane]
    const v16bf* sBlo = (const v16bf*)sXlo;
    const v16bf* Ahi  = (const v16bf*)Whi;
    const v16bf* Alo  = (const v16bf*)Wlo;

    float bestv = -3.4e38f;
    int   besti = 0;

    auto upd = [&](const v8f& acc, int mtl) {
        const int jbase = mtl * 16 + half * 8;
        #pragma unroll
        for (int v = 0; v < 8; ++v) {
            float val = acc[v] + bpts[p * PRED_LEN + jbase + v];
            if (val > bestv) { bestv = val; besti = jbase + v; }  // strict: keep lowest idx
        }
    };

    const int gtbase = p * MT_PER_P;
    const int bbase  = wv * (KCHUNKS * 32);

    // 10 pairs of M-tiles: two independent accumulator chains share B loads.
    for (int mt0 = 0; mt0 < MT_PER_P - 1; mt0 += 2) {
        const int g0 = (gtbase + mt0) * 16;
        const int g1 = g0 + 16;
        v8f acc0 = {0.f, 0.f, 0.f, 0.f, 0.f, 0.f, 0.f, 0.f};
        v8f acc1 = {0.f, 0.f, 0.f, 0.f, 0.f, 0.f, 0.f, 0.f};
        #pragma unroll 2
        for (int kc = 0; kc < KCHUNKS; ++kc) {
            v16bf bh  = sBhi[bbase + kc * 32 + lane];
            v16bf bl  = sBlo[bbase + kc * 32 + lane];
            v16bf ah0 = Ahi[(g0 + kc) * 32 + lane];
            v16bf al0 = Alo[(g0 + kc) * 32 + lane];
            v16bf ah1 = Ahi[(g1 + kc) * 32 + lane];
            v16bf al1 = Alo[(g1 + kc) * 32 + lane];
            // bf16x3 fp32 emulation: hi*hi + hi*lo + lo*hi, fp32 accumulate
            acc0 = __builtin_amdgcn_wmma_f32_16x16x32_bf16(
                       false, ah0, false, bh, (short)0, acc0, false, false);
            acc1 = __builtin_amdgcn_wmma_f32_16x16x32_bf16(
                       false, ah1, false, bh, (short)0, acc1, false, false);
            acc0 = __builtin_amdgcn_wmma_f32_16x16x32_bf16(
                       false, ah0, false, bl, (short)0, acc0, false, false);
            acc1 = __builtin_amdgcn_wmma_f32_16x16x32_bf16(
                       false, ah1, false, bl, (short)0, acc1, false, false);
            acc0 = __builtin_amdgcn_wmma_f32_16x16x32_bf16(
                       false, al0, false, bh, (short)0, acc0, false, false);
            acc1 = __builtin_amdgcn_wmma_f32_16x16x32_bf16(
                       false, al1, false, bh, (short)0, acc1, false, false);
        }
        upd(acc0, mt0);
        upd(acc1, mt0 + 1);
    }
    // Tail M-tile (mt = 20).
    {
        const int g0 = (gtbase + MT_PER_P - 1) * 16;
        v8f acc = {0.f, 0.f, 0.f, 0.f, 0.f, 0.f, 0.f, 0.f};
        #pragma unroll 2
        for (int kc = 0; kc < KCHUNKS; ++kc) {
            v16bf bh = sBhi[bbase + kc * 32 + lane];
            v16bf bl = sBlo[bbase + kc * 32 + lane];
            v16bf ah = Ahi[(g0 + kc) * 32 + lane];
            v16bf al = Alo[(g0 + kc) * 32 + lane];
            acc = __builtin_amdgcn_wmma_f32_16x16x32_bf16(
                      false, ah, false, bh, (short)0, acc, false, false);
            acc = __builtin_amdgcn_wmma_f32_16x16x32_bf16(
                      false, ah, false, bl, (short)0, acc, false, false);
            acc = __builtin_amdgcn_wmma_f32_16x16x32_bf16(
                      false, al, false, bh, (short)0, acc, false, false);
        }
        upd(acc, MT_PER_P - 1);
    }

    // Cross-half combine (lane L holds M rows 0-7 best, L+16 holds M rows 8-15).
    float ov = __shfl_xor(bestv, 16);
    int   oi = __shfl_xor(besti, 16);
    if ((ov > bestv) || (ov == bestv && oi < besti)) { bestv = ov; besti = oi; }

    if (lane < 16) pts[(bg * BR + wv * 16 + lane) * N_PTS + p] = besti;
}

// ---------------------------------------------------------------------------
// Tiny params GEMM: (8192 x 512) @ (512 x 18), plain fp32.
// ---------------------------------------------------------------------------
__global__ __launch_bounds__(256) void params_kernel(
    const float* __restrict__ x,
    const float* __restrict__ Wc,
    const float* __restrict__ bc,
    float* __restrict__ params)
{
    int idx = blockIdx.x * 256 + threadIdx.x;
    if (idx >= BATCH * N_SLOTS * 2) return;
    int b = idx / (N_SLOTS * 2);
    int o = idx % (N_SLOTS * 2);
    const float* xr = x  + b * SEQ_LEN;
    const float* wr = Wc + o * SEQ_LEN;
    float last = xr[SEQ_LEN - 1];
    float acc = 0.f;
    for (int k = 0; k < SEQ_LEN; ++k)
        acc = fmaf(xr[k] - last, wr[k], acc);
    params[idx] = acc + bc[o];
}

// ---------------------------------------------------------------------------
// Compose: slot(t) = #{i: pts[i] <= t}; out = a*t + b + last_val.
// ---------------------------------------------------------------------------
__global__ __launch_bounds__(256) void finalize_kernel(
    const float* __restrict__ x,
    const int*   __restrict__ pts,
    const float* __restrict__ params,
    float* __restrict__ out)
{
    int idx = blockIdx.x * 256 + threadIdx.x;
    if (idx >= BATCH * PRED_LEN) return;
    int b = idx / PRED_LEN;
    int t = idx % PRED_LEN;
    const int* pp = pts + b * N_PTS;
    int c = 0;
    #pragma unroll
    for (int i = 0; i < N_PTS; ++i) c += (pp[i] <= t) ? 1 : 0;
    float a  = params[b * (N_SLOTS * 2) + c * 2 + 0];
    float bb = params[b * (N_SLOTS * 2) + c * 2 + 1];
    out[idx] = fmaf(a, (float)t, bb) + x[b * SEQ_LEN + (SEQ_LEN - 1)];
}

// ---------------------------------------------------------------------------
extern "C" void kernel_launch(void* const* d_in, const int* in_sizes, int n_in,
                              void* d_out, int out_size, void* d_ws, size_t ws_size,
                              hipStream_t stream)
{
    const float* x      = (const float*)d_in[0];
    const float* W_pts  = (const float*)d_in[1];
    const float* b_pts  = (const float*)d_in[2];
    const float* W_comp = (const float*)d_in[3];
    const float* b_comp = (const float*)d_in[4];
    float* out = (float*)d_out;

    // ws partition (all offsets 32B-aligned); total ~6.4 MB
    char* ws = (char*)d_ws;
    const size_t whi_bytes = (size_t)W_SW_ELEMS * 2;          // 2,752,512
    __bf16* Whi    = (__bf16*)(ws);
    __bf16* Wlo    = (__bf16*)(ws + whi_bytes);
    int*    pts    = (int*)   (ws + 2 * whi_bytes);           // 8192*8 ints
    float*  params = (float*) (ws + 2 * whi_bytes + (size_t)BATCH * N_PTS * 4);

    // 1) swizzle W into WMMA A-matrix layout (bf16 hi/lo)
    prep_w_kernel<<<(W_SW_ELEMS + 255) / 256, 256, 0, stream>>>(W_pts, Whi, Wlo);

    // 2) main GEMM + fused argmax: grid (batch groups, point heads)
    dim3 ggrid(BATCH / BR, N_PTS);
    gemm_argmax_kernel<<<ggrid, GEMM_THREADS, 0, stream>>>(x, Whi, Wlo, b_pts, pts);

    // 3) params GEMM
    params_kernel<<<(BATCH * N_SLOTS * 2 + 255) / 256, 256, 0, stream>>>(
        x, W_comp, b_comp, params);

    // 4) compose output
    finalize_kernel<<<(BATCH * PRED_LEN + 255) / 256, 256, 0, stream>>>(
        x, pts, params, out);
}
